// RNNClassifier_76459007804085
// MI455X (gfx1250) — compile-verified, matrix operand order checked by reading, not run on hardware
//
#include <hip/hip_runtime.h>
#include <hip/hip_bf16.h>

// ---------------------------------------------------------------------------
// Phased-LSTM (2 layers, B=256, T=2048, F=3, U=256, NCLASS=5) for gfx1250.
// One workgroup per 16-row batch tile; f16 WMMA (16x16x32, f32 accum) for the
// recurrent GEMMs; h/c state in registers; weights pre-swizzled into WMMA
// B-fragment layout so each fragment is one contiguous 32B-per-lane load.
// kt-outer / j-inner loop order: 8 independent accumulator chains (no WMMA
// RAW back-to-back) and only one live A fragment. An opaque per-iteration
// zero offset stops LICM from hoisting (and spilling) the weight stream.
// ---------------------------------------------------------------------------

typedef __attribute__((ext_vector_type(16))) _Float16 v16h;
typedef __attribute__((ext_vector_type(8)))  float    v8f;

#define T_STEPS 2048
#define UNITS   256
#define GATES   1024     // 4*U
#define NT      64       // GATES/16 n-tiles
#define R_ON    0.05f
#define ALPHA   0.001f

// ---------------------------------------------------------------------------
// Prologue: convert fp32 weights into f16 WMMA B-fragment tiles.
// Logical matrix is K x 1024 where rows k<256 come from srcA, rows >=256 from
// srcB (used to concatenate [W1;U1] into one K=512 operand).
// Fragment storage: frag = nt*KT + kt, then 32 lanes x 16 halves contiguous,
// matching the 16-bit operand layout (VGPR v holds K pair; lane<16 = K-low
// half, lane>=16 = K-high half; lane&15 = N column).
// ---------------------------------------------------------------------------
__global__ void plstm_tile_weights(const float* __restrict__ srcA,
                                   const float* __restrict__ srcB,
                                   _Float16* __restrict__ dst, int KT) {
  int idx = blockIdx.x * 256 + threadIdx.x;
  int total = NT * KT * 512;
  if (idx >= total) return;
  int slot = idx & 15;
  int lane = (idx >> 4) & 31;
  int frag = idx >> 9;
  int kt = frag % KT;
  int nt = frag / KT;
  int v = slot >> 1, half = slot & 1;
  int klocal = ((v < 4) ? 0 : 16) + ((lane < 16) ? 0 : 8) + (v & 3) * 2 + half;
  int k = kt * 32 + klocal;
  int n = nt * 16 + (lane & 15);
  float val = (k < 256) ? srcA[k * GATES + n] : srcB[(k - 256) * GATES + n];
  dst[idx] = (_Float16)val;
}

// Load a 16x32 f16 A-fragment from LDS (row-major h buffer, row stride 512).
// The K-pair pattern makes the 8 dword reads two contiguous 16B groups, which
// the compiler merges into 2x ds_load_b128.
__device__ __forceinline__ v16h load_a(const _Float16* hA, int m, int koff,
                                       int kcol0) {
  union { v16h h; unsigned u[8]; } r;
  const unsigned* row = (const unsigned*)(hA + m * 512);
  #pragma unroll
  for (int v = 0; v < 8; ++v) {
    int kb = kcol0 + ((v < 4) ? 0 : 16) + (v & 3) * 2 + koff;  // even
    r.u[v] = row[kb >> 1];
  }
  return r.h;
}

__device__ __forceinline__ float sigmoidf_(float x) {
  return 1.0f / (1.0f + __expf(-x));
}

__device__ __forceinline__ float timegate(float t, float s, float tau) {
  float q = (t - s) / tau;
  float phi = q - floorf(q);                 // jnp.mod(., tau)/tau
  return (phi < 0.5f * R_ON) ? (2.0f / R_ON) * phi
       : (phi < R_ON)        ? 2.0f - (2.0f / R_ON) * phi
                             : ALPHA * phi;
}

// ---------------------------------------------------------------------------
// Main persistent kernel: grid = 16 blocks x 256 threads (8 waves).
// ---------------------------------------------------------------------------
__global__ __launch_bounds__(256, 1) void plstm_kernel(
    const float* __restrict__ inputs,  // [B, T, 3]
    const float* __restrict__ times,   // [B, T]
    const float* __restrict__ W0,      // [3, 1024] fp32
    const float* __restrict__ b0, const float* __restrict__ tau0,
    const float* __restrict__ s0, const float* __restrict__ b1,
    const float* __restrict__ tau1, const float* __restrict__ s1,
    const float* __restrict__ Wfc,     // [256, 5]
    const float* __restrict__ bfc,     // [5]
    const _Float16* __restrict__ u0t,  // U0 tiled, KT=8
    const _Float16* __restrict__ w1t,  // [W1;U1] tiled, KT=16
    float* __restrict__ out)           // [B, T, 5]
{
  __shared__ __align__(16) _Float16 hA[16 * 512];  // rows: [h0(256) | h1(256)]
  __shared__ float h1f[16 * 256];
  __shared__ float wfcl[UNITS * 5];
  __shared__ float xl[16 * 3];
  __shared__ float tl[16];
  __shared__ float zfc[16 * 5];

  const int tid  = threadIdx.x;
  const int wave = tid >> 5;
  const int l    = tid & 31;
  const int c    = l & 15;                 // N column / A row within tile
  const int hoff = (l < 16) ? 0 : 8;       // C/D: M = r + hoff
  const int koff = (l < 16) ? 0 : 8;       // A/B: K-half select
  const int row0 = blockIdx.x * 16;

  for (int i = tid; i < 16 * 512; i += 256) hA[i] = (_Float16)0.0f;
  for (int i = tid; i < UNITS * 5; i += 256) wfcl[i] = Wfc[i];
  __syncthreads();

  // Per-lane constants: each lane owns columns ncol(j)=16*(wave+8j)+c.
  float w0a[8], w0b[8], w0c[8], b0r[8], b1r[8];
  #pragma unroll
  for (int j = 0; j < 8; ++j) {
    int ncol = 16 * (wave + 8 * j) + c;
    w0a[j] = W0[ncol];
    w0b[j] = W0[GATES + ncol];
    w0c[j] = W0[2 * GATES + ncol];
    b0r[j] = b0[ncol];
    b1r[j] = b1[ncol];
  }
  const int u0i = 16 * wave + c;           // unit owned at p=0
  const int u1i = u0i + 128;               // unit owned at p=1
  float tau0v[2] = {tau0[u0i], tau0[u1i]}, s0v[2] = {s0[u0i], s0[u1i]};
  float tau1v[2] = {tau1[u0i], tau1[u1i]}, s1v[2] = {s1[u0i], s1[u1i]};

  float c0r[2][8] = {}, h0r[2][8] = {}, c1r[2][8] = {}, h1r[2][8] = {};

  for (int step = 0; step < T_STEPS; ++step) {
    // Opaque zero: makes weight addresses loop-variant so the B-fragment
    // stream stays inside the loop as global_load_b128 (no LICM + spill).
    int zoff;
    asm volatile("s_mov_b32 %0, 0" : "=s"(zoff));
    const _Float16* u0s = u0t + zoff;
    const _Float16* w1s = w1t + zoff;

    if (tid < 16) tl[tid] = times[(row0 + tid) * T_STEPS + step];
    if (tid < 48) {
      int m = tid / 3, f = tid % 3;
      xl[m * 3 + f] = inputs[((row0 + m) * T_STEPS + step) * 3 + f];
    }
    __syncthreads();

    // ================= layer 0: z = h0 @ U0 + b0 + x @ W0 =================
    v8f accs[8];
    #pragma unroll
    for (int j = 0; j < 8; ++j)
      #pragma unroll
      for (int r = 0; r < 8; ++r) accs[j][r] = b0r[j];

    __builtin_prefetch(u0s + (unsigned)l * 16, 0, 1);
    #pragma unroll
    for (int kt = 0; kt < 8; ++kt) {
      v16h a = load_a(hA, c, koff, kt * 32);
      #pragma unroll
      for (int j = 0; j < 8; ++j) {
        const int nt = wave + 8 * j;
        v16h bf = *(const v16h*)(u0s + (((nt * 8 + kt) * 32) + l) * 16);
        accs[j] = __builtin_amdgcn_wmma_f32_16x16x32_f16(
            false, a, false, bf, (short)0, accs[j], false, false);
      }
    }
    #pragma unroll
    for (int j = 0; j < 8; ++j)
      #pragma unroll
      for (int r = 0; r < 8; ++r) {
        int m = r + hoff;
        accs[j][r] += xl[m * 3] * w0a[j] + xl[m * 3 + 1] * w0b[j] +
                      xl[m * 3 + 2] * w0c[j];
      }
    __syncthreads();  // all waves done reading hA h0 region

    // gates, phased time gate, h0/c0 update (slots: i=0+p f=2+p g=4+p o=6+p)
    #pragma unroll
    for (int p = 0; p < 2; ++p) {
      const int u = (p == 0) ? u0i : u1i;
      #pragma unroll
      for (int r = 0; r < 8; ++r) {
        int m = r + hoff;
        float iv = sigmoidf_(accs[0 + p][r]);
        float fv = sigmoidf_(accs[2 + p][r]);
        float gv = tanhf(accs[4 + p][r]);
        float ov = sigmoidf_(accs[6 + p][r]);
        float cn = fv * c0r[p][r] + iv * gv;
        float hn = ov * tanhf(cn);
        float k = timegate(tl[m], s0v[p], tau0v[p]);
        c0r[p][r] = k * cn + (1.0f - k) * c0r[p][r];
        h0r[p][r] = k * hn + (1.0f - k) * h0r[p][r];
        hA[m * 512 + u] = (_Float16)h0r[p][r];
      }
    }
    __syncthreads();  // h0 visible

    // ============ layer 1: z = [h0,h1] @ [W1;U1] + b1 (K = 512) ============
    #pragma unroll
    for (int j = 0; j < 8; ++j)
      #pragma unroll
      for (int r = 0; r < 8; ++r) accs[j][r] = b1r[j];

    __builtin_prefetch(w1s + (unsigned)l * 16, 0, 1);
    #pragma unroll
    for (int kt = 0; kt < 16; ++kt) {
      v16h a = load_a(hA, c, koff, kt * 32);   // cols 0..511 = [h0 | h1]
      #pragma unroll
      for (int j = 0; j < 8; ++j) {
        const int nt = wave + 8 * j;
        v16h bf = *(const v16h*)(w1s + (((nt * 16 + kt) * 32) + l) * 16);
        accs[j] = __builtin_amdgcn_wmma_f32_16x16x32_f16(
            false, a, false, bf, (short)0, accs[j], false, false);
      }
    }
    __syncthreads();  // all waves done reading hA h1 region

    #pragma unroll
    for (int p = 0; p < 2; ++p) {
      const int u = (p == 0) ? u0i : u1i;
      #pragma unroll
      for (int r = 0; r < 8; ++r) {
        int m = r + hoff;
        float iv = sigmoidf_(accs[0 + p][r]);
        float fv = sigmoidf_(accs[2 + p][r]);
        float gv = tanhf(accs[4 + p][r]);
        float ov = sigmoidf_(accs[6 + p][r]);
        float cn = fv * c1r[p][r] + iv * gv;
        float hn = ov * tanhf(cn);
        float k = timegate(tl[m], s1v[p], tau1v[p]);
        c1r[p][r] = k * cn + (1.0f - k) * c1r[p][r];
        h1r[p][r] = k * hn + (1.0f - k) * h1r[p][r];
        hA[m * 512 + 256 + u] = (_Float16)h1r[p][r];
        h1f[m * 256 + u] = h1r[p][r];
      }
    }
    __syncthreads();  // h1 visible

    // ===================== classifier head + softmax ======================
    if (tid < 80) {
      int m = tid / 5, cls = tid % 5;
      float acc = bfc[cls];
      for (int u = 0; u < UNITS; ++u) acc += h1f[m * 256 + u] * wfcl[u * 5 + cls];
      zfc[m * 5 + cls] = acc;
    }
    __syncthreads();
    if (tid < 16) {
      int m = tid;
      float mx = zfc[m * 5];
      #pragma unroll
      for (int cc = 1; cc < 5; ++cc) mx = fmaxf(mx, zfc[m * 5 + cc]);
      float e[5], sum = 0.0f;
      #pragma unroll
      for (int cc = 0; cc < 5; ++cc) { e[cc] = __expf(zfc[m * 5 + cc] - mx); sum += e[cc]; }
      float inv = 1.0f / sum;
      #pragma unroll
      for (int cc = 0; cc < 5; ++cc)
        out[((size_t)(row0 + m) * T_STEPS + step) * 5 + cc] = e[cc] * inv;
    }
    __syncthreads();
  }
}

extern "C" void kernel_launch(void* const* d_in, const int* in_sizes, int n_in,
                              void* d_out, int out_size, void* d_ws, size_t ws_size,
                              hipStream_t stream) {
  const float* inputs = (const float*)d_in[0];
  const float* times  = (const float*)d_in[1];
  const float* W0     = (const float*)d_in[2];
  const float* U0     = (const float*)d_in[3];
  const float* b0     = (const float*)d_in[4];
  const float* tau0   = (const float*)d_in[5];
  const float* s0     = (const float*)d_in[6];
  const float* W1     = (const float*)d_in[7];
  const float* U1     = (const float*)d_in[8];
  const float* b1     = (const float*)d_in[9];
  const float* tau1   = (const float*)d_in[10];
  const float* s1     = (const float*)d_in[11];
  const float* Wfc    = (const float*)d_in[12];
  const float* bfc    = (const float*)d_in[13];

  _Float16* u0t = (_Float16*)d_ws;                       // 256*1024 halves
  _Float16* w1t = (_Float16*)d_ws + 256 * 1024;          // 512*1024 halves

  int totU0 = NT * 8 * 512;
  int totW1 = NT * 16 * 512;
  plstm_tile_weights<<<(totU0 + 255) / 256, 256, 0, stream>>>(U0, U0, u0t, 8);
  plstm_tile_weights<<<(totW1 + 255) / 256, 256, 0, stream>>>(W1, U1, w1t, 16);
  plstm_kernel<<<16, 256, 0, stream>>>(inputs, times, W0, b0, tau0, s0,
                                       b1, tau1, s1, Wfc, bfc, u0t, w1t,
                                       (float*)d_out);
}
